// ProtoNet_44684839747671
// MI455X (gfx1250) — compile-verified, matrix operand order checked by reading, not run on hardware
//
#include <hip/hip_runtime.h>
#include <hip/hip_bf16.h>

// ---------------------------------------------------------------------------
// Problem constants
// ---------------------------------------------------------------------------
#define BATCH   2
#define HH      256
#define WW      256
#define NPIX    (BATCH * HH * WW)     // 131072 pixels
#define C0      720                   // true channel count
#define CP      736                   // padded to 23*32 for WMMA K loop
#define KSEG    19
#define MPROT   10
#define KM      190                   // real proto rows
#define KMP     192                   // padded to 12*16 col tiles

// GEMM tiling: 8 waves/block, each wave owns a 32(M) x 48(N) tile.
#define GWAVES  8
#define WM      32                    // rows per wave   (2 m-tiles)
#define WN      48                    // cols per wave   (3 n-tiles; 48 | 720 and 48 | 192)
#define BM      (GWAVES * WM)         // 256 rows per block

typedef unsigned short u16;
typedef __attribute__((ext_vector_type(16))) __bf16 v16bf;
typedef __attribute__((ext_vector_type(8)))  float  v8f;

union FragBF {
    v16bf v;
    uint4 q[2];
};

__device__ __forceinline__ u16 f32_to_bf16_rne(float f) {
    union { float f; unsigned u; } x;
    x.f = f;
    unsigned u = x.u;
    u += 0x7fffu + ((u >> 16) & 1u);   // round-to-nearest-even
    return (u16)(u >> 16);
}

// align_corners=True bilinear sample of a square SxS map at output pixel (h,w) of 256x256
__device__ __forceinline__ float bilin_ac(const float* __restrict__ f, int b, int c,
                                          int C, int S, int h, int w) {
    float sc = (float)(S - 1) / 255.0f;
    float ys = h * sc, xs = w * sc;
    int y0 = (int)floorf(ys), x0 = (int)floorf(xs);
    int y1 = min(y0 + 1, S - 1), x1 = min(x0 + 1, S - 1);
    float wy = ys - (float)y0, wx = xs - (float)x0;
    const float* base = f + ((long)b * C + c) * (long)S * S;
    float p00 = base[(long)y0 * S + x0], p01 = base[(long)y0 * S + x1];
    float p10 = base[(long)y1 * S + x0], p11 = base[(long)y1 * S + x1];
    return (1.f - wy) * ((1.f - wx) * p00 + wx * p01) +
           wy        * ((1.f - wx) * p10 + wx * p11);
}

// ---------------------------------------------------------------------------
// 1) Pack a [720,720] fp32 weight matrix into bf16 [720, CP] (zero K-padding)
// ---------------------------------------------------------------------------
__global__ void pack_weight_kernel(const float* __restrict__ w, u16* __restrict__ out) {
    int o = blockIdx.x;                       // 720 blocks
    for (int c = threadIdx.x; c < CP; c += blockDim.x) {
        float v = (c < C0) ? w[(long)o * C0 + c] : 0.f;
        out[(long)o * CP + c] = f32_to_bf16_rne(v);
    }
}

// ---------------------------------------------------------------------------
// 2) L2-normalize prototypes over C, pack bf16 [KMP, CP] (pad rows/cols zero)
// ---------------------------------------------------------------------------
__global__ void pack_protos_kernel(const float* __restrict__ p, u16* __restrict__ out) {
    __shared__ float red[8];
    __shared__ float bcast;
    int r = blockIdx.x;                       // KMP blocks, 256 threads
    float vals[3];
    float s = 0.f;
    if (r < KM) {
        for (int j = 0; j < 3; ++j) {
            int c = threadIdx.x + j * 256;
            float v = (c < C0) ? p[(long)r * C0 + c] : 0.f;
            vals[j] = v;
            s += v * v;
        }
    } else {
        vals[0] = vals[1] = vals[2] = 0.f;
    }
    for (int off = 16; off > 0; off >>= 1) s += __shfl_xor(s, off, 32);
    if ((threadIdx.x & 31) == 0) red[threadIdx.x >> 5] = s;
    __syncthreads();
    if (threadIdx.x == 0) {
        float t = 0.f;
        for (int i = 0; i < 8; ++i) t += red[i];
        bcast = 1.f / fmaxf(sqrtf(t), 1e-12f);
    }
    __syncthreads();
    float inv = (r < KM) ? bcast : 0.f;
    for (int j = 0; j < 3; ++j) {
        int c = threadIdx.x + j * 256;
        if (c < CP) out[(long)r * CP + c] = f32_to_bf16_rne(vals[j] * inv);
    }
}

// ---------------------------------------------------------------------------
// 3) Build concat(feat1, up(feat2), up(feat3), up(feat4)) as bf16 [NPIX, CP]
// ---------------------------------------------------------------------------
__global__ void build_feats_kernel(const float* __restrict__ f1, const float* __restrict__ f2,
                                   const float* __restrict__ f3, const float* __restrict__ f4,
                                   u16* __restrict__ feats) {
    int n = blockIdx.x;                       // NPIX blocks
    int b = n >> 16;
    int h = (n >> 8) & 255;
    int w = n & 255;
    for (int c = threadIdx.x; c < CP; c += blockDim.x) {
        float v;
        if (c < 48)       v = f1[(((long)b * 48 + c) * HH + h) * WW + w];
        else if (c < 144) v = bilin_ac(f2, b, c - 48, 96, 128, h, w);
        else if (c < 336) v = bilin_ac(f3, b, c - 144, 192, 64, h, w);
        else if (c < C0)  v = bilin_ac(f4, b, c - 336, 384, 32, h, w);
        else              v = 0.f;
        feats[(long)n * CP + c] = f32_to_bf16_rne(v);
    }
}

// ---------------------------------------------------------------------------
// 4) bf16 WMMA GEMM:  out[n, o] = sum_k A[n,k] * B[o,k]  (+bias, opt. relu)
//    A: [rows, lda] bf16 row-major, B: [cols, ldb] bf16 row-major.
//    Per-wave 32x48 tile (2x3 WMMA tiles): A fragments reused 3x, B fragments
//    reused 2x -> 1.67 b128 loads per v_wmma. grid.x = column groups so that
//    blocks sharing an A slab are launch-adjacent (A streams from HBM once;
//    weights stay L2-resident).
// ---------------------------------------------------------------------------
__global__ void gemm_wmma_bf16(const u16* __restrict__ A, const u16* __restrict__ B,
                               int lda, int ldb, int K,
                               const float* __restrict__ bias, int relu,
                               u16* __restrict__ out_bf, float* __restrict__ out_f32,
                               int ldc) {
    const int lane = threadIdx.x & 31;
    const int wv   = threadIdx.x >> 5;
    const long row0 = (long)blockIdx.y * BM + wv * WM;   // 32 rows for this wave
    const int  col0 = blockIdx.x * WN;                   // 48 cols for this wave
    const int  r  = lane & 15;
    const int  hi = lane >> 4;

    // 16-bit A-matrix fragment layout (16x32):
    //   lanes 0-15 row r hold K {kk..kk+7, kk+16..kk+23};
    //   lanes 16-31 row r hold K {kk+8..kk+15, kk+24..kk+31}.
    const u16* arow0 = A + (row0 + r) * (long)lda + hi * 8;
    const u16* arow1 = arow0 + 16 * (long)lda;
    // B-matrix column o == row o of weights; same K packing per lane.
    const u16* brow0 = B + (long)(col0 + r) * ldb + hi * 8;
    const u16* brow1 = brow0 + 16 * (long)ldb;
    const u16* brow2 = brow0 + 32 * (long)ldb;

    v8f acc00 = {}, acc01 = {}, acc02 = {};
    v8f acc10 = {}, acc11 = {}, acc12 = {};

    for (int kk = 0; kk < K; kk += 32) {
        FragBF a0, a1, b0, b1, b2;
        a0.q[0] = *(const uint4*)(arow0 + kk);
        a0.q[1] = *(const uint4*)(arow0 + kk + 16);
        a1.q[0] = *(const uint4*)(arow1 + kk);
        a1.q[1] = *(const uint4*)(arow1 + kk + 16);
        b0.q[0] = *(const uint4*)(brow0 + kk);
        b0.q[1] = *(const uint4*)(brow0 + kk + 16);
        b1.q[0] = *(const uint4*)(brow1 + kk);
        b1.q[1] = *(const uint4*)(brow1 + kk + 16);
        b2.q[0] = *(const uint4*)(brow2 + kk);
        b2.q[1] = *(const uint4*)(brow2 + kk + 16);

        acc00 = __builtin_amdgcn_wmma_f32_16x16x32_bf16(false, a0.v, false, b0.v, (short)0, acc00, false, false);
        acc01 = __builtin_amdgcn_wmma_f32_16x16x32_bf16(false, a0.v, false, b1.v, (short)0, acc01, false, false);
        acc02 = __builtin_amdgcn_wmma_f32_16x16x32_bf16(false, a0.v, false, b2.v, (short)0, acc02, false, false);
        acc10 = __builtin_amdgcn_wmma_f32_16x16x32_bf16(false, a1.v, false, b0.v, (short)0, acc10, false, false);
        acc11 = __builtin_amdgcn_wmma_f32_16x16x32_bf16(false, a1.v, false, b1.v, (short)0, acc11, false, false);
        acc12 = __builtin_amdgcn_wmma_f32_16x16x32_bf16(false, a1.v, false, b2.v, (short)0, acc12, false, false);
    }

    // D layout: VGPR v -> M = v + 8*hi, N = lane&15
    v8f accs[2][3] = { { acc00, acc01, acc02 }, { acc10, acc11, acc12 } };
    for (int i = 0; i < 2; ++i) {
        for (int t = 0; t < 3; ++t) {
            const int colw = col0 + t * 16 + r;
            const float bv = bias ? bias[colw] : 0.f;
            for (int v = 0; v < 8; ++v) {
                long rw = row0 + i * 16 + v + 8 * hi;
                float val = accs[i][t][v] + bv;
                if (relu) val = fmaxf(val, 0.f);
                if (out_bf) out_bf[rw * (long)ldc + colw] = f32_to_bf16_rne(val);
                else        out_f32[rw * (long)ldc + colw] = val;
            }
        }
    }
}

// ---------------------------------------------------------------------------
// 5) Zero the K-padding columns [C0, CP) of a bf16 [NPIX, CP] buffer
// ---------------------------------------------------------------------------
__global__ void zero_pad_kernel(u16* __restrict__ buf) {
    long idx = (long)blockIdx.x * blockDim.x + threadIdx.x;   // NPIX*16 threads
    long row = idx >> 4;
    int  col = C0 + (int)(idx & 15);
    buf[row * CP + col] = 0;
}

// ---------------------------------------------------------------------------
// 6) Per-(b,h,o) sum of squares over width -> 1/max(||.||, eps)
// ---------------------------------------------------------------------------
__global__ void width_sumsq_kernel(const float* __restrict__ x2, float* __restrict__ invw) {
    int bh = blockIdx.x;                      // BATCH*HH blocks
    for (int o = threadIdx.x; o < C0; o += blockDim.x) {
        const float* base = x2 + ((long)bh * WW) * C0 + o;
        float s = 0.f;
        for (int w = 0; w < WW; ++w) {
            float v = base[(long)w * C0];
            s += v * v;
        }
        invw[(long)bh * C0 + o] = 1.f / fmaxf(sqrtf(s), 1e-12f);
    }
}

// ---------------------------------------------------------------------------
// 7) Fused per-pixel: width-L2 scale -> LayerNorm(C) -> L2(C) -> bf16 [NPIX,CP]
// ---------------------------------------------------------------------------
__global__ void rownorm_kernel(const float* __restrict__ x2, const float* __restrict__ invw,
                               const float* __restrict__ g, const float* __restrict__ bta,
                               u16* __restrict__ cout) {
    __shared__ float red[8];
    __shared__ float bcast;
    int n = blockIdx.x;                       // NPIX blocks, 256 threads
    int bh = n >> 8;

    auto block_sum = [&](float v) -> float {
        for (int off = 16; off > 0; off >>= 1) v += __shfl_xor(v, off, 32);
        if ((threadIdx.x & 31) == 0) red[threadIdx.x >> 5] = v;
        __syncthreads();
        if (threadIdx.x == 0) {
            float t = 0.f;
            for (int i = 0; i < 8; ++i) t += red[i];
            bcast = t;
        }
        __syncthreads();
        float rv = bcast;
        __syncthreads();
        return rv;
    };

    float vals[3];
    float s = 0.f;
    for (int j = 0; j < 3; ++j) {
        int c = threadIdx.x + j * 256;
        float v = 0.f;
        if (c < C0) v = x2[(long)n * C0 + c] * invw[(long)bh * C0 + c];
        vals[j] = v;
        s += v;
    }
    float mean = block_sum(s) * (1.0f / C0);

    float sq = 0.f;
    for (int j = 0; j < 3; ++j) {
        int c = threadIdx.x + j * 256;
        if (c < C0) { float d = vals[j] - mean; sq += d * d; }
    }
    float var = block_sum(sq) * (1.0f / C0);
    float rstd = rsqrtf(var + 1e-5f);

    float l2 = 0.f;
    for (int j = 0; j < 3; ++j) {
        int c = threadIdx.x + j * 256;
        if (c < C0) {
            float y = (vals[j] - mean) * rstd * g[c] + bta[c];
            vals[j] = y;
            l2 += y * y;
        }
    }
    float inv = 1.f / fmaxf(sqrtf(block_sum(l2)), 1e-12f);

    for (int j = 0; j < 3; ++j) {
        int c = threadIdx.x + j * 256;
        if (c < C0)      cout[(long)n * CP + c] = f32_to_bf16_rne(vals[j] * inv);
        else if (c < CP) cout[(long)n * CP + c] = 0;
    }
}

// ---------------------------------------------------------------------------
// 8) max over M prototypes, LayerNorm over K=19, write NCHW output
// ---------------------------------------------------------------------------
__global__ void final_kernel(const float* __restrict__ masks,
                             const float* __restrict__ mg, const float* __restrict__ mb,
                             float* __restrict__ out) {
    long n = (long)blockIdx.x * blockDim.x + threadIdx.x;
    if (n >= NPIX) return;
    const float* mrow = masks + n * KMP;
    float s[KSEG];
    float mean = 0.f;
    for (int k = 0; k < KSEG; ++k) {
        float mx = mrow[k * MPROT];
        for (int m = 1; m < MPROT; ++m) mx = fmaxf(mx, mrow[k * MPROT + m]);
        s[k] = mx;
        mean += mx;
    }
    mean *= (1.0f / KSEG);
    float var = 0.f;
    for (int k = 0; k < KSEG; ++k) { float d = s[k] - mean; var += d * d; }
    var *= (1.0f / KSEG);
    float rstd = rsqrtf(var + 1e-5f);
    int b = (int)(n >> 16), h = (int)((n >> 8) & 255), w = (int)(n & 255);
    for (int k = 0; k < KSEG; ++k) {
        float v = (s[k] - mean) * rstd * mg[k] + mb[k];
        out[(((long)b * KSEG + k) * HH + h) * WW + w] = v;
    }
}

// ---------------------------------------------------------------------------
// Launch
// ---------------------------------------------------------------------------
extern "C" void kernel_launch(void* const* d_in, const int* in_sizes, int n_in,
                              void* d_out, int out_size, void* d_ws, size_t ws_size,
                              hipStream_t stream) {
    const float* feat1 = (const float*)d_in[0];
    const float* feat2 = (const float*)d_in[1];
    const float* feat3 = (const float*)d_in[2];
    const float* feat4 = (const float*)d_in[3];
    const float* pw1   = (const float*)d_in[4];
    const float* pb1   = (const float*)d_in[5];
    const float* pw2   = (const float*)d_in[6];
    const float* pb2   = (const float*)d_in[7];
    const float* fng   = (const float*)d_in[8];
    const float* fnb   = (const float*)d_in[9];
    const float* mng   = (const float*)d_in[10];
    const float* mnb   = (const float*)d_in[11];
    const float* prot  = (const float*)d_in[12];
    float* out = (float*)d_out;

    // Workspace layout (256B aligned)
    char* ws = (char*)d_ws;
    const size_t OFF_W1P   = 0;                                   // 720*CP bf16
    const size_t OFF_W2P   = OFF_W1P + (size_t)C0 * CP * 2;
    const size_t OFF_PROT  = OFF_W2P + (size_t)C0 * CP * 2;       // KMP*CP bf16
    const size_t OFF_INVW  = OFF_PROT + (size_t)KMP * CP * 2;     // BATCH*HH*C0 f32
    const size_t OFF_FEATS = OFF_INVW + (size_t)BATCH * HH * C0 * 4;  // NPIX*CP bf16 (reused as _c)
    const size_t OFF_Y     = OFF_FEATS + (size_t)NPIX * CP * 2;       // NPIX*CP bf16 (reused as masks f32)
    const size_t OFF_X2    = OFF_Y + (size_t)NPIX * CP * 2;           // NPIX*C0 f32

    u16*   w1p   = (u16*)(ws + OFF_W1P);
    u16*   w2p   = (u16*)(ws + OFF_W2P);
    u16*   protp = (u16*)(ws + OFF_PROT);
    float* invw  = (float*)(ws + OFF_INVW);
    u16*   feats = (u16*)(ws + OFF_FEATS);
    u16*   cbuf  = (u16*)(ws + OFF_FEATS);    // _c reuses feats region
    u16*   ybuf  = (u16*)(ws + OFF_Y);
    float* masks = (float*)(ws + OFF_Y);      // masks reuses Y region
    float* x2    = (float*)(ws + OFF_X2);

    // 1) pack weights / protos
    pack_weight_kernel<<<C0, 256, 0, stream>>>(pw1, w1p);
    pack_weight_kernel<<<C0, 256, 0, stream>>>(pw2, w2p);
    pack_protos_kernel<<<KMP, 256, 0, stream>>>(prot, protp);

    // 2) build concatenated/upsampled features (bf16, K-padded)
    build_feats_kernel<<<NPIX, 256, 0, stream>>>(feat1, feat2, feat3, feat4, feats);

    // 3) GEMM1: Y = relu(feats * W1^T + b1)   (bf16 out, [NPIX, CP])
    {
        dim3 grid(C0 / WN, NPIX / BM);        // 15 x 512 (cols fastest -> A slab reuse)
        gemm_wmma_bf16<<<grid, 256, 0, stream>>>(feats, w1p, CP, CP, CP,
                                                 pb1, 1, ybuf, nullptr, CP);
    }
    zero_pad_kernel<<<(NPIX * 16) / 256, 256, 0, stream>>>(ybuf);

    // 4) GEMM2: X2 = Y * W2^T + b2            (f32 out, [NPIX, C0])
    {
        dim3 grid(C0 / WN, NPIX / BM);
        gemm_wmma_bf16<<<grid, 256, 0, stream>>>(ybuf, w2p, CP, CP, CP,
                                                 pb2, 0, nullptr, x2, C0);
    }

    // 5) width-wise L2 stats, then fused width-scale + LN + L2 -> _c (bf16)
    width_sumsq_kernel<<<BATCH * HH, 256, 0, stream>>>(x2, invw);
    rownorm_kernel<<<NPIX, 256, 0, stream>>>(x2, invw, fng, fnb, cbuf);

    // 6) GEMM3: masks = _c * protos_norm^T    (f32 out, [NPIX, KMP])
    {
        dim3 grid(KMP / WN, NPIX / BM);       // 4 x 512
        gemm_wmma_bf16<<<grid, 256, 0, stream>>>(cbuf, protp, CP, CP, CP,
                                                 nullptr, 0, nullptr, masks, KMP);
    }

    // 7) max over M, LayerNorm over K, NCHW transpose
    final_kernel<<<NPIX / 256, 256, 0, stream>>>(masks, mng, mnb, out);
}
